// Encoder_2207613190309
// MI455X (gfx1250) — matrix operand order repaired and checked by reading, not compiled, not run on hardware
//
#include <hip/hip_runtime.h>

// ---------------- problem constants ----------------
#define BB 8
#define TT 1024
#define CC 512
#define FCC 2048
#define HH 8
#define KCC 64
#define LL 6

typedef __attribute__((ext_vector_type(16))) __bf16 v16bf;
typedef __attribute__((ext_vector_type(8)))  float  v8f;
typedef __attribute__((ext_vector_type(4)))  unsigned int u32x4;

union Frag { unsigned int u[8]; u32x4 q4[2]; v16bf v; };

__device__ __forceinline__ unsigned short f2bf(float f) {
  unsigned int u = __float_as_uint(f);
  unsigned int r = u + 0x7FFFu + ((u >> 16) & 1u);   // round-to-nearest-even
  return (unsigned short)(r >> 16);
}
__device__ __forceinline__ float bf2f(unsigned short h) {
  return __uint_as_float(((unsigned int)h) << 16);
}
__device__ __forceinline__ unsigned int pack2(unsigned short lo, unsigned short hi) {
  return (unsigned int)lo | ((unsigned int)hi << 16);
}
__device__ __forceinline__ v8f wmma_bf16(const Frag& a, const Frag& b, v8f c) {
  return __builtin_amdgcn_wmma_f32_16x16x32_bf16(false, a.v, false, b.v,
                                                 (short)0, c, false, false);
}

// A fragment (16x32 bf16): rowBase points at [row][chunk_k0]; row stride must be 16B-mult.
__device__ __forceinline__ void load_afrag(Frag& f, const unsigned short* rowBase, int lh) {
  const u32x4* p = (const u32x4*)(rowBase + lh * 8);
  f.q4[0] = p[0];        // K = lh*8 .. lh*8+7
  f.q4[1] = p[2];        // K = 16+lh*8 .. 16+lh*8+7  (+32 bytes)
}
// B fragment (32x16 bf16) from an [N][K] tile: rowBase points at [n][chunk_k0].
__device__ __forceinline__ void load_bfrag(Frag& f, const unsigned short* rowBase, int lh) {
  const u32x4* p = (const u32x4*)(rowBase + lh * 8);
  f.q4[0] = p[0];
  f.q4[1] = p[2];
}

// =====================================================================
// Generic GEMM / ks-3 conv with double-buffered LDS staging:
//   Y[b,o,t] = act((sum_{kk,c} W[o,c,kk] * X[b,c,t+kk-off] + bias[o]) * ascale)
// 128 threads = 4 waves (2x2); block tile 64(M=o) x 64(N=t);
// each wave: 32x32 = 2 A-frags x 2 B-frags x 4 WMMAs per 32-K chunk.
// =====================================================================
template<int KSZ, bool XBF16, bool RELU, bool OUTBF16>
__global__ __launch_bounds__(128)
void gemm_conv(const float* __restrict__ W, const void* __restrict__ Xv,
               const float* __restrict__ bias, void* __restrict__ Yv,
               int Cout, int Cin, float ascale)
{
  const int tbase = blockIdx.x * 64;
  const int obase = blockIdx.y * 64;
  const int b     = blockIdx.z;
  const int tid   = threadIdx.x;
  const int wave  = tid >> 5, lane = tid & 31;
  const int lr    = lane & 15, lh = lane >> 4;
  const int wm    = wave >> 1, wn = wave & 1;

  __shared__ __align__(16) unsigned short a_lds[2][64][40];  // [o][c] bf16 (80B rows)
  __shared__ __align__(16) unsigned short x_lds[2][64][40];  // [t][c] bf16 (transposed)

  const float*          Xf = (const float*)Xv;
  const unsigned short* Xh = (const unsigned short*)Xv;

  const int CH  = Cin / 32;
  const int NCH = KSZ * CH;

  auto stage = [&](int u, int bi) {
    int kk = u / CH;
    int cb = (u - kk * CH) * 32;
    int shift = kk - (KSZ - 1) / 2;            // KSZ=1 -> 0; KSZ=3 -> -1,0,1
    for (int i = 0; i < 8; ++i) {              // weights 64x32 as [o][c]
      int idx = tid + i * 128;
      int ol = idx >> 4, cp = idx & 15;        // c = 2*cp
      size_t w0 = ((size_t)(obase + ol) * Cin + (cb + 2 * cp)) * KSZ + kk;
      *(unsigned int*)&a_lds[bi][ol][2 * cp] = pack2(f2bf(W[w0]), f2bf(W[w0 + KSZ]));
    }
    for (int i = 0; i < 8; ++i) {              // X 32x64 transposed to [t][c]
      int idx = tid + i * 128;
      int cp = idx >> 6, n = idx & 63;
      int t  = tbase + n + shift;
      unsigned int pv = 0;
      if (t >= 0 && t < TT) {
        size_t off = ((size_t)b * Cin + (cb + 2 * cp)) * TT + t;
        pv = XBF16 ? pack2(Xh[off], Xh[off + TT])
                   : pack2(f2bf(Xf[off]), f2bf(Xf[off + TT]));
      }
      *(unsigned int*)&x_lds[bi][n][2 * cp] = pv;
    }
  };

  v8f acc[4];
  for (int q = 0; q < 4; ++q)
    for (int r = 0; r < 8; ++r) acc[q][r] = 0.0f;

  stage(0, 0);
  __syncthreads();
  for (int u = 0; u < NCH; ++u) {
    const int bi = u & 1;
    if (u + 1 < NCH) stage(u + 1, bi ^ 1);     // overlap next-chunk staging with WMMAs
    Frag fa[2], fb[2];
    load_afrag(fa[0], &a_lds[bi][wm * 32 + lr][0], lh);
    load_afrag(fa[1], &a_lds[bi][wm * 32 + 16 + lr][0], lh);
    load_bfrag(fb[0], &x_lds[bi][wn * 32 + lr][0], lh);
    load_bfrag(fb[1], &x_lds[bi][wn * 32 + 16 + lr][0], lh);
    acc[0] = wmma_bf16(fa[0], fb[0], acc[0]);
    acc[1] = wmma_bf16(fa[0], fb[1], acc[1]);
    acc[2] = wmma_bf16(fa[1], fb[0], acc[2]);
    acc[3] = wmma_bf16(fa[1], fb[1], acc[3]);
    __syncthreads();
  }

  // epilogue: D element (M,N): lane<16 -> M=r ; lane>=16 -> M=8+r ; N=lr
  float*          Yf = (float*)Yv;
  unsigned short* Yh = (unsigned short*)Yv;
  for (int i = 0; i < 2; ++i) {
    for (int j = 0; j < 2; ++j) {
      int t = tbase + wn * 32 + j * 16 + lr;
      for (int r = 0; r < 8; ++r) {
        int o = obase + wm * 32 + i * 16 + lh * 8 + r;
        float val = (acc[i * 2 + j][r] + bias[o]) * ascale;
        if (RELU) val = fmaxf(val, 0.0f);
        size_t off = ((size_t)b * Cout + o) * TT + t;
        if (OUTBF16) Yh[off] = f2bf(val); else Yf[off] = val;
      }
    }
  }
}

// =====================================================================
// Flash attention with banded relative-position terms.
// One block = one (b,h) x 64-query tile. 128 threads = 4 waves (16x64 strips).
// O accumulator kept in register fragments (WMMA C-operand accumulation);
// per-row flash rescale applied in registers via alpha[64] in LDS.
// scores[t,s] = q[t].k[s] + (|s-t|<=4 ? q[t].erk[s-t+4] : 0)   (mask all-ones)
// out[t]     = (O[t] + sum_j band[t][j]*erv[j]) / l[t]
// =====================================================================
__global__ __launch_bounds__(128)
void attn_kernel(const float* __restrict__ Q, const float* __restrict__ K,
                 const float* __restrict__ V, const float* __restrict__ erk,
                 const float* __restrict__ erv, float* __restrict__ O)
{
  const int qb   = blockIdx.x * 64;
  const int bh   = blockIdx.y;
  const int b    = bh >> 3, h = bh & 7;
  const int tid  = threadIdx.x;
  const int wave = tid >> 5, lane = tid & 31;
  const int lr   = lane & 15, lh = lane >> 4;

  const size_t base = ((size_t)b * CC + h * KCC) * TT;
  const float* qp = Q + base;
  const float* kp = K + base;
  const float* vp = V + base;

  __shared__ __align__(16) unsigned short q_tile[64][72];  // [t][d] bf16 (144B rows)
  __shared__ __align__(16) unsigned short k_tile[64][72];  // [s][d]  (transposed)
  __shared__ __align__(16) unsigned short v_tile[64][72];  // [d][s]  (natural copy)
  __shared__ __align__(16) float s_tile[64][68];           // scores; P bf16 overlays; O dump
  __shared__ float qrel[64][9];
  __shared__ float band[64][9];
  __shared__ __align__(16) float a_row[64];
  __shared__ float m_row[64], l_row[64];

  // stage Q transposed -> [t][d], paired b32 stores
  for (int i = 0; i < 16; ++i) {
    int idx = tid + i * 128;                 // 0..2047
    int dp = idx >> 6, tl = idx & 63;        // d = 2*dp
    size_t off = (size_t)(2 * dp) * TT + qb + tl;
    *(unsigned int*)&q_tile[tl][2 * dp] = pack2(f2bf(qp[off]), f2bf(qp[off + TT]));
  }
  __syncthreads();

  // persistent Q fragments (2 K-chunks of 32)
  Frag aq[2];
  {
    int row = wave * 16 + lr;
    load_afrag(aq[0], &q_tile[row][0], lh);
    load_afrag(aq[1], &q_tile[row][32], lh);
  }
  // qrel[t][j] = q[t] . erk[j]  (b32 paired reads of Q)
  for (int e = tid; e < 576; e += 128) {
    int t = e / 9, j = e % 9;
    const unsigned int* qrow = (const unsigned int*)&q_tile[t][0];
    float s = 0.0f;
    for (int dp = 0; dp < 32; ++dp) {
      unsigned int pv = qrow[dp];
      s += bf2f((unsigned short)pv)         * erk[j * 64 + 2 * dp]
         + bf2f((unsigned short)(pv >> 16)) * erk[j * 64 + 2 * dp + 1];
    }
    qrel[t][j] = s;
  }
  for (int i = tid; i < 64; i += 128) { m_row[i] = -3.0e38f; l_row[i] = 0.0f; }
  for (int i = tid; i < 576; i += 128) (&band[0][0])[i] = 0.0f;
  __syncthreads();

  v8f oacc[4];                               // O fragments: rows=queries, cols=d
  for (int q = 0; q < 4; ++q)
    for (int r = 0; r < 8; ++r) oacc[q][r] = 0.0f;

  for (int kb = 0; kb < TT / 64; ++kb) {
    const int sb = kb * 64;
    // K transposed -> [s][d] (paired over d); V natural -> [d][s] (float2 over s)
    for (int i = 0; i < 16; ++i) {
      int idx = tid + i * 128;
      int dp = idx >> 6, sl = idx & 63;
      size_t off = (size_t)(2 * dp) * TT + sb + sl;
      *(unsigned int*)&k_tile[sl][2 * dp] = pack2(f2bf(kp[off]), f2bf(kp[off + TT]));
    }
    for (int i = 0; i < 16; ++i) {
      int idx = tid + i * 128;
      int d = idx >> 5, sp = idx & 31;
      const float2* vv = (const float2*)(vp + (size_t)d * TT + sb);
      float2 w = vv[sp];
      *(unsigned int*)&v_tile[d][2 * sp] = pack2(f2bf(w.x), f2bf(w.y));
    }
    __syncthreads();

    // S = Q * K^T   (B-frags from [s][d] tile)
    for (int nt = 0; nt < 4; ++nt) {
      v8f acc; for (int r = 0; r < 8; ++r) acc[r] = 0.0f;
      for (int c = 0; c < 2; ++c) {
        Frag fb;
        load_bfrag(fb, &k_tile[nt * 16 + lr][c * 32], lh);
        acc = wmma_bf16(aq[c], fb, acc);
      }
      int n = nt * 16 + lr;
      for (int r = 0; r < 8; ++r)
        s_tile[wave * 16 + lh * 8 + r][n] = acc[r];
    }
    __syncthreads();

    // online softmax per row (64 row-threads)
    if (tid < 64) {
      int t = tid, tg = qb + t;
      float mx = m_row[t], nm = mx;
      for (int s = 0; s < 64; ++s) {
        float sv = s_tile[t][s];
        int j = sb + s - tg;
        if (j >= -4 && j <= 4) sv += qrel[t][j + 4];
        s_tile[t][s] = sv;
        nm = fmaxf(nm, sv);
      }
      float alpha = __expf(mx - nm);
      float l = l_row[t] * alpha;
      for (int j = 0; j < 9; ++j) band[t][j] *= alpha;
      unsigned short* pr = (unsigned short*)&s_tile[t][0];  // P overlays S in-place
      for (int s = 0; s < 64; ++s) {
        float e = __expf(s_tile[t][s] - nm);
        l += e;
        int jj = sb + s - tg + 4;
        if (jj >= 0 && jj < 9) band[t][jj] += e;
        pr[s] = f2bf(e);   // byte 2s overlaps float s/2 (<= s, already consumed)
      }
      m_row[t] = nm; l_row[t] = l;
      a_row[t] = alpha;
    }
    __syncthreads();

    // rescale O fragments in registers: each lane holds rows wave*16+lh*8 .. +7
    {
      const float4 al0 = *(const float4*)&a_row[wave * 16 + lh * 8];
      const float4 al1 = *(const float4*)&a_row[wave * 16 + lh * 8 + 4];
      const float ar[8] = {al0.x, al0.y, al0.z, al0.w, al1.x, al1.y, al1.z, al1.w};
      for (int nt = 0; nt < 4; ++nt)
        for (int r = 0; r < 8; ++r) oacc[nt][r] *= ar[r];
    }
    // O += P * V directly into register fragments (WMMA C-accumulation)
    Frag pa[2];
    {
      const unsigned short* pb = (const unsigned short*)&s_tile[0][0];
      int row = wave * 16 + lr;
      load_afrag(pa[0], pb + row * 136, lh);       // 272B rows: 16B-mult
      load_afrag(pa[1], pb + row * 136 + 32, lh);
    }
    for (int nt = 0; nt < 4; ++nt) {
      Frag fb0, fb1;
      load_bfrag(fb0, &v_tile[nt * 16 + lr][0], lh);
      load_bfrag(fb1, &v_tile[nt * 16 + lr][32], lh);
      oacc[nt] = wmma_bf16(pa[1], fb1, wmma_bf16(pa[0], fb0, oacc[nt]));
    }
    __syncthreads();
  }

  // dump O fragments to LDS once, then banded-erv finalize
  for (int nt = 0; nt < 4; ++nt) {
    int n = nt * 16 + lr;
    for (int r = 0; r < 8; ++r)
      s_tile[wave * 16 + lh * 8 + r][n] = oacc[nt][r];
  }
  __syncthreads();
  for (int e = tid; e < 64 * 64; e += 128) {
    int t = e >> 6, d = e & 63;
    float val = s_tile[t][d];
    for (int j = 0; j < 9; ++j) val += band[t][j] * erv[j * 64 + d];
    O[((size_t)b * CC + h * KCC + d) * TT + qb + t] = val / l_row[t];
  }
}

// =====================================================================
// Fused residual-add + channel LayerNorm: out = LN(X + Y) over c of [b,c,t]
// =====================================================================
__global__ __launch_bounds__(256)
void ln_kernel(const float* __restrict__ X, const float* __restrict__ Y,
               const float* __restrict__ g, const float* __restrict__ bt,
               float* __restrict__ out)
{
  int t = blockIdx.x * 256 + threadIdx.x;
  int b = blockIdx.y;
  size_t colbase = (size_t)b * CC * TT + t;
  float s = 0.0f, ss = 0.0f;
  for (int c = 0; c < CC; ++c) {
    size_t off = colbase + (size_t)c * TT;
    float v = X[off] + Y[off];
    s += v; ss += v * v;
  }
  float m  = s / CC;
  float vr = ss / CC - m * m;
  float rs = rsqrtf(vr + 1e-5f);
  for (int c = 0; c < CC; ++c) {
    size_t off = colbase + (size_t)c * TT;
    float v = X[off] + Y[off];
    out[off] = (v - m) * rs * g[c] + bt[c];
  }
}

__global__ __launch_bounds__(256)
void mask_mul(const float* __restrict__ x, const float* __restrict__ mask,
              float* __restrict__ out)
{
  int i = blockIdx.x * 256 + threadIdx.x;
  int t = i % TT;
  int b = i / (CC * TT);
  out[i] = x[i] * mask[b * TT + t];
}

// =====================================================================
extern "C" void kernel_launch(void* const* d_in, const int* in_sizes, int n_in,
                              void* d_out, int out_size, void* d_ws, size_t ws_size,
                              hipStream_t stream) {
  (void)in_sizes; (void)n_in; (void)out_size; (void)ws_size;
  const float* x      = (const float*)d_in[0];
  const float* x_mask = (const float*)d_in[1];
  const float* qw = (const float*)d_in[2];
  const float* qb = (const float*)d_in[3];
  const float* kw = (const float*)d_in[4];
  const float* kb = (const float*)d_in[5];
  const float* vw = (const float*)d_in[6];
  const float* vb = (const float*)d_in[7];
  const float* ow = (const float*)d_in[8];
  const float* ob = (const float*)d_in[9];
  const float* erk = (const float*)d_in[10];
  const float* erv = (const float*)d_in[11];
  const float* ln1g = (const float*)d_in[12];
  const float* ln1b = (const float*)d_in[13];
  const float* fw1 = (const float*)d_in[14];
  const float* fb1 = (const float*)d_in[15];
  const float* fw2 = (const float*)d_in[16];
  const float* fb2 = (const float*)d_in[17];
  const float* ln2g = (const float*)d_in[18];
  const float* ln2b = (const float*)d_in[19];

  const size_t S = (size_t)BB * CC * TT;        // 4,194,304 elems = 16 MB f32
  char* ws = (char*)d_ws;
  float* xb   = (float*)(ws + 0 * S * 4);
  float* qbuf = (float*)(ws + 1 * S * 4);
  float* kbuf = (float*)(ws + 2 * S * 4);
  float* vbuf = (float*)(ws + 3 * S * 4);
  float* abuf = (float*)(ws + 4 * S * 4);
  float* ybuf = (float*)(ws + 5 * S * 4);
  unsigned short* h1 = (unsigned short*)(ws + 1 * S * 4);  // bf16 [B,FC,T]; aliases dead q/k/v

  const float qscale = 1.0f / 8.0f;             // 1/sqrt(KC)
  dim3 blkE(256), grdE((unsigned)(S / 256));
  dim3 ggC(TT / 64, CC / 64, BB);
  dim3 ggF(TT / 64, FCC / 64, BB);
  dim3 gA(TT / 64, BB * HH);
  dim3 gLN(TT / 256, BB);

  mask_mul<<<grdE, blkE, 0, stream>>>(x, x_mask, xb);

  for (int i = 0; i < LL; ++i) {
    const float* qwi = qw + (size_t)i * CC * CC;
    const float* kwi = kw + (size_t)i * CC * CC;
    const float* vwi = vw + (size_t)i * CC * CC;
    const float* owi = ow + (size_t)i * CC * CC;
    const float* fw1i = fw1 + (size_t)i * FCC * CC * 3;
    const float* fw2i = fw2 + (size_t)i * CC * FCC * 3;

    gemm_conv<1, false, false, false><<<ggC, 128, 0, stream>>>(qwi, xb, qb + i * CC, qbuf, CC, CC, qscale);
    gemm_conv<1, false, false, false><<<ggC, 128, 0, stream>>>(kwi, xb, kb + i * CC, kbuf, CC, CC, 1.0f);
    gemm_conv<1, false, false, false><<<ggC, 128, 0, stream>>>(vwi, xb, vb + i * CC, vbuf, CC, CC, 1.0f);

    attn_kernel<<<gA, 128, 0, stream>>>(qbuf, kbuf, vbuf,
                                        erk + (size_t)i * 9 * KCC,
                                        erv + (size_t)i * 9 * KCC, abuf);

    gemm_conv<1, false, false, false><<<ggC, 128, 0, stream>>>(owi, abuf, ob + i * CC, ybuf, CC, CC, 1.0f);
    ln_kernel<<<gLN, 256, 0, stream>>>(xb, ybuf, ln1g + i * CC, ln1b + i * CC, xb);

    gemm_conv<3, false, true,  true ><<<ggF, 128, 0, stream>>>(fw1i, xb, fb1 + i * FCC, h1, FCC, CC, 1.0f);
    gemm_conv<3, true,  false, false><<<ggC, 128, 0, stream>>>(fw2i, h1, fb2 + i * CC, ybuf, CC, FCC, 1.0f);
    ln_kernel<<<gLN, 256, 0, stream>>>(xb, ybuf, ln2g + i * CC, ln2b + i * CC, xb);
  }

  mask_mul<<<grdE, blkE, 0, stream>>>(xb, x_mask, (float*)d_out);
}
